// OPTAttention_10402410791606
// MI455X (gfx1250) — matrix operand order repaired and checked by reading, not compiled, not run on hardware
//
#include <hip/hip_runtime.h>
#include <hip/hip_bf16.h>

typedef __attribute__((ext_vector_type(16))) __bf16 v16bf;
typedef __attribute__((ext_vector_type(8)))  float  v8f;

#define EMBED     2048
#define NUM_HEADS 32
#define HEAD_DIM  64
#define SCALING   0.125f
#define NEGF      (-3.4028234663852886e+38f)

union FragAB { v16bf v; uint4 q[2]; unsigned short s[16]; };
union FragC  { v8f   v; float f[8]; };

__device__ __forceinline__ unsigned short f32_to_bf16_rne(float x) {
  unsigned u = __float_as_uint(x);
  unsigned r = 0x7FFFu + ((u >> 16) & 1u);
  return (unsigned short)((u + r) >> 16);
}

// ---- DPP-based 16-lane reductions (no LDS, no dscnt waits) ----------------
// wave32: DPP "row" = 16 lanes, so lanes 0-15 / 16-31 reduce independently,
// matching the WMMA C-layout half-groups.
template<int CTRL>
__device__ __forceinline__ float dpp_mov_f32(float x) {
  int i = __float_as_int(x);
  int r = __builtin_amdgcn_update_dpp(i, i, CTRL, 0xf, 0xf, false);
  return __int_as_float(r);
}
__device__ __forceinline__ float row16_max(float x) {
  x = fmaxf(x, dpp_mov_f32<0xB1>(x));    // quad_perm [1,0,3,2]  (xor 1)
  x = fmaxf(x, dpp_mov_f32<0x4E>(x));    // quad_perm [2,3,0,1]  (xor 2)
  x = fmaxf(x, dpp_mov_f32<0x141>(x));   // ROW_HALF_MIRROR      (4s -> 8s)
  x = fmaxf(x, dpp_mov_f32<0x140>(x));   // ROW_MIRROR           (8s -> 16)
  return x;
}
__device__ __forceinline__ float row16_sum(float x) {
  x += dpp_mov_f32<0xB1>(x);
  x += dpp_mov_f32<0x4E>(x);
  x += dpp_mov_f32<0x141>(x);
  x += dpp_mov_f32<0x140>(x);
  return x;
}

// ---------------------------------------------------------------------------
// fp32 -> bf16 conversion
// ---------------------------------------------------------------------------
__global__ __launch_bounds__(256) void cvt_f32_bf16(const float* __restrict__ in,
                                                    unsigned short* __restrict__ out,
                                                    int n) {
  int i = blockIdx.x * blockDim.x + threadIdx.x;
  if (i < n) out[i] = f32_to_bf16_rne(in[i]);
}

// ---------------------------------------------------------------------------
// GEMM:  C[M,N] = A[M,K](bf16) @ W[N,K]^T(bf16) + bias, all WMMA bf16->f32.
//   MODE 0: out = bf16, scattered to [B,H,T,D] head layout     (Q, K)
//   MODE 1: out = f32 row-major [M,N]                          (final proj)
//   MODE 2: out = bf16, scattered to [B,H,D,T] (transposed V for P@V B-frags)
// Block = 256 thr (8 waves); block tile 128x64; wave tile 32x32 (2x2 WMMA).
// ---------------------------------------------------------------------------
template<int MODE>
__global__ __launch_bounds__(256)
void gemm_wmma_bf16(const unsigned short* __restrict__ A,
                    const unsigned short* __restrict__ W,
                    const float* __restrict__ bias, float scale,
                    unsigned short* __restrict__ outBf,
                    float* __restrict__ outF,
                    int M, int N, int K, int T, int H, int D)
{
  const int lane = threadIdx.x & 31;
  const int w    = threadIdx.x >> 5;
  const int g    = lane >> 4;     // lane half-group (0/1)
  const int ln   = lane & 15;
  const int mBase = blockIdx.y * 128 + (w >> 1) * 32;
  const int nBase = blockIdx.x * 64  + (w & 1)  * 32;

  FragC acc[2][2];
  #pragma unroll
  for (int i = 0; i < 2; ++i)
    #pragma unroll
    for (int j = 0; j < 2; ++j)
      #pragma unroll
      for (int r = 0; r < 8; ++r) acc[i][j].f[r] = 0.f;

  for (int kb = 0; kb < K; kb += 32) {
    if (kb + 32 < K)   // pull next A stripe toward L2/L0 (global_prefetch_b8)
      __builtin_prefetch(A + (size_t)(mBase + ln) * K + kb + 32, 0, 0);

    FragAB a[2], b[2];
    #pragma unroll
    for (int mt = 0; mt < 2; ++mt) {   // A 16x32: lane=row, k = g*16 + h (contig)
      const unsigned short* p = A + (size_t)(mBase + mt * 16 + ln) * K + kb + g * 16;
      a[mt].q[0] = *(const uint4*)(p);
      a[mt].q[1] = *(const uint4*)(p + 8);
    }
    #pragma unroll
    for (int nt = 0; nt < 2; ++nt) {   // B 32x16: lane=col n, k = g*16 + h (contig)
      const unsigned short* p = W + (size_t)(nBase + nt * 16 + ln) * K + kb + g * 16;
      b[nt].q[0] = *(const uint4*)(p);
      b[nt].q[1] = *(const uint4*)(p + 8);
    }
    #pragma unroll
    for (int mt = 0; mt < 2; ++mt)
      #pragma unroll
      for (int nt = 0; nt < 2; ++nt)
        acc[mt][nt].v = __builtin_amdgcn_wmma_f32_16x16x32_bf16(
            false, a[mt].v, false, b[nt].v, (short)0, acc[mt][nt].v, false, false);
  }

  #pragma unroll
  for (int mt = 0; mt < 2; ++mt)
    #pragma unroll
    for (int nt = 0; nt < 2; ++nt)
      #pragma unroll
      for (int r = 0; r < 8; ++r) {
        const int m = mBase + mt * 16 + r + 8 * g;   // C layout: row = vgpr + 8*group
        const int n = nBase + nt * 16 + ln;
        const float v = (acc[mt][nt].f[r] + bias[n]) * scale;
        if (MODE == 1) {
          outF[(size_t)m * N + n] = v;
        } else {
          const int bI = m / T, t = m % T, h = n / D, d = n % D;
          const size_t head = (size_t)(bI * H + h);
          const size_t idx = (MODE == 0) ? (head * T + t) * D + d
                                         : (head * D + d) * T + t;   // V transposed
          outBf[idx] = f32_to_bf16_rne(v);
        }
      }
}

// ---------------------------------------------------------------------------
// Causal flash attention. Q pre-scaled. Q/K: bf16 [B*H, T, D]; V: bf16
// [B*H, D, T] (transposed so the P@V B-fragment reads contiguous keys).
// One wave owns a 16-row Q tile; 8 waves/block share one (b,h).
// Per 32-key step: 4 WMMAs for S = Q K^T, online softmax (DPP 16-lane
// reductions), P f32->bf16 through LDS (C-layout -> A-layout), 4 WMMAs P@V.
// ctx out: bf16 [B, T, H, D]  (== row-major [B*T, E]).
// ---------------------------------------------------------------------------
__global__ __launch_bounds__(256)
void flash_attn_wmma(const unsigned short* __restrict__ Q,
                     const unsigned short* __restrict__ Km,
                     const unsigned short* __restrict__ Vt,
                     unsigned short* __restrict__ ctx,
                     int T, int H, int D)
{
  __shared__ unsigned short ldsP[8][16 * 32];
  const int lane = threadIdx.x & 31;
  const int w    = threadIdx.x >> 5;
  const int g    = lane >> 4;
  const int ln   = lane & 15;
  const int bh   = blockIdx.y;
  const int q0   = blockIdx.x * 128 + w * 16;

  const unsigned short* Qh = Q  + (size_t)bh * T * D;
  const unsigned short* Kh = Km + (size_t)bh * T * D;
  const unsigned short* Vh = Vt + (size_t)bh * T * D;   // [D, T]

  FragAB qa[2];                       // 16 x 64 Q tile = two k-steps
  #pragma unroll
  for (int ks = 0; ks < 2; ++ks) {
    const unsigned short* p = Qh + (size_t)(q0 + ln) * D + ks * 32 + g * 16;
    qa[ks].q[0] = *(const uint4*)(p);
    qa[ks].q[1] = *(const uint4*)(p + 8);
  }

  FragC acc[4];
  #pragma unroll
  for (int dt = 0; dt < 4; ++dt)
    #pragma unroll
    for (int r = 0; r < 8; ++r) acc[dt].f[r] = 0.f;
  float rmax[8], rsum[8];
  #pragma unroll
  for (int r = 0; r < 8; ++r) { rmax[r] = NEGF; rsum[r] = 0.f; }

  for (int s0 = 0; s0 <= q0 + 15; s0 += 32) {
    // prefetch next key block's K stripe while this one computes
    if (s0 + 32 <= q0 + 15)
      __builtin_prefetch(Kh + (size_t)(s0 + 32 + ln) * D, 0, 0);

    // ---- scores: two 16x16 tiles over keys [s0, s0+32) -------------------
    FragC S[2];
    #pragma unroll
    for (int c = 0; c < 2; ++c) {
      #pragma unroll
      for (int r = 0; r < 8; ++r) S[c].f[r] = 0.f;
      #pragma unroll
      for (int ks = 0; ks < 2; ++ks) {
        FragAB kf;   // B-frag: lane = key col, k = d contiguous from K[key][...]
        const unsigned short* p = Kh + (size_t)(s0 + c * 16 + ln) * D + ks * 32 + g * 16;
        kf.q[0] = *(const uint4*)(p);
        kf.q[1] = *(const uint4*)(p + 8);
        S[c].v = __builtin_amdgcn_wmma_f32_16x16x32_bf16(
            false, qa[ks].v, false, kf.v, (short)0, S[c].v, false, false);
      }
    }
    // ---- causal mask + online softmax (DPP reductions) -------------------
    #pragma unroll
    for (int r = 0; r < 8; ++r) {
      const int mg = q0 + r + 8 * g;
      const float v0 = ((s0 + ln)      <= mg) ? S[0].f[r] : NEGF;
      const float v1 = ((s0 + 16 + ln) <= mg) ? S[1].f[r] : NEGF;
      const float mb   = row16_max(fmaxf(v0, v1));
      const float mnew = fmaxf(rmax[r], mb);
      const float corr = __expf(rmax[r] - mnew);
      rmax[r] = mnew;
      const float p0 = __expf(v0 - mnew);
      const float p1 = __expf(v1 - mnew);
      const float sb = row16_sum(p0 + p1);
      rsum[r] = rsum[r] * corr + sb;
      #pragma unroll
      for (int dt = 0; dt < 4; ++dt) acc[dt].f[r] *= corr;
      const int row = r + 8 * g;
      ldsP[w][row * 32 + ln]      = f32_to_bf16_rne(p0);
      ldsP[w][row * 32 + 16 + ln] = f32_to_bf16_rne(p1);
    }
    asm volatile("s_wait_dscnt 0" ::: "memory");   // LDS C->A layout handoff
    FragAB pa;                                     // P as A-frag 16x32
    {
      const unsigned short* p = &ldsP[w][ln * 32 + g * 16];
      pa.q[0] = *(const uint4*)(p);
      pa.q[1] = *(const uint4*)(p + 8);
    }
    // ---- ctx += P @ V  (V transposed: contiguous keys per lane) ----------
    #pragma unroll
    for (int dt = 0; dt < 4; ++dt) {
      FragAB vf;   // B-frag: lane = d col, k = key = g*16 + h contiguous in Vt
      const unsigned short* p = Vh + (size_t)(dt * 16 + ln) * T + s0 + g * 16;
      vf.q[0] = *(const uint4*)(p);
      vf.q[1] = *(const uint4*)(p + 8);
      acc[dt].v = __builtin_amdgcn_wmma_f32_16x16x32_bf16(
          false, pa.v, false, vf.v, (short)0, acc[dt].v, false, false);
    }
  }

  const int bI = bh / H, h = bh % H;
  #pragma unroll
  for (int dt = 0; dt < 4; ++dt)
    #pragma unroll
    for (int r = 0; r < 8; ++r) {
      const int m = q0 + r + 8 * g;
      const int d = dt * 16 + ln;
      ctx[((size_t)(bI * T + m) * H + h) * D + d] =
          f32_to_bf16_rne(acc[dt].f[r] / rsum[r]);
    }
}

// ---------------------------------------------------------------------------
extern "C" void kernel_launch(void* const* d_in, const int* in_sizes, int n_in,
                              void* d_out, int out_size, void* d_ws, size_t ws_size,
                              hipStream_t stream) {
  const int B = 2, T = 2048, E = EMBED, H = NUM_HEADS, D = HEAD_DIM;
  const int M = B * T;            // 4096
  (void)in_sizes; (void)n_in; (void)out_size; (void)ws_size;

  const float* x  = (const float*)d_in[0];
  const float* Wq = (const float*)d_in[1];
  const float* bq = (const float*)d_in[2];
  const float* Wk = (const float*)d_in[3];
  const float* bk = (const float*)d_in[4];
  const float* Wv = (const float*)d_in[5];
  const float* bv = (const float*)d_in[6];
  const float* Wo = (const float*)d_in[7];
  const float* bo = (const float*)d_in[8];

  size_t off = 0;
  auto alloc = [&](size_t elems) {
    unsigned short* p = (unsigned short*)((char*)d_ws + off);
    off += ((elems * 2) + 255) & ~(size_t)255;
    return p;
  };
  unsigned short* Xbf  = alloc((size_t)M * E);
  unsigned short* Wqbf = alloc((size_t)E * E);
  unsigned short* Wkbf = alloc((size_t)E * E);
  unsigned short* Wvbf = alloc((size_t)E * E);
  unsigned short* Wobf = alloc((size_t)E * E);
  unsigned short* Qbf  = alloc((size_t)M * E);   // [B*H, T, D]
  unsigned short* Kbf  = alloc((size_t)M * E);   // [B*H, T, D]
  unsigned short* Vtbf = alloc((size_t)M * E);   // [B*H, D, T]  (transposed)
  unsigned short* Cbf  = alloc((size_t)M * E);   // [B, T, H, D] == [M, E]

  // 1) convert fp32 -> bf16
  {
    const int nx = M * E, nw = E * E;
    cvt_f32_bf16<<<(nx + 255) / 256, 256, 0, stream>>>(x,  Xbf,  nx);
    cvt_f32_bf16<<<(nw + 255) / 256, 256, 0, stream>>>(Wq, Wqbf, nw);
    cvt_f32_bf16<<<(nw + 255) / 256, 256, 0, stream>>>(Wk, Wkbf, nw);
    cvt_f32_bf16<<<(nw + 255) / 256, 256, 0, stream>>>(Wv, Wvbf, nw);
    cvt_f32_bf16<<<(nw + 255) / 256, 256, 0, stream>>>(Wo, Wobf, nw);
  }

  // 2) Q/K/V projections (WMMA); V written transposed for the P@V B-frags
  dim3 gGemm(E / 64, M / 128);
  gemm_wmma_bf16<0><<<gGemm, 256, 0, stream>>>(Xbf, Wqbf, bq, SCALING, Qbf, nullptr,
                                               M, E, E, T, H, D);
  gemm_wmma_bf16<0><<<gGemm, 256, 0, stream>>>(Xbf, Wkbf, bk, 1.0f, Kbf, nullptr,
                                               M, E, E, T, H, D);
  gemm_wmma_bf16<2><<<gGemm, 256, 0, stream>>>(Xbf, Wvbf, bv, 1.0f, Vtbf, nullptr,
                                               M, E, E, T, H, D);

  // 3) fused causal flash attention (WMMA)
  dim3 gAttn(T / 128, B * H);
  flash_attn_wmma<<<gAttn, 256, 0, stream>>>(Qbf, Kbf, Vtbf, Cbf, T, H, D);

  // 4) output projection -> fp32 d_out
  gemm_wmma_bf16<1><<<gGemm, 256, 0, stream>>>(Cbf, Wobf, bo, 1.0f, nullptr,
                                               (float*)d_out, M, E, E, T, H, D);
}
